// DMFM_Lite_22024592293913
// MI455X (gfx1250) — compile-verified, matrix operand order checked by reading, not run on hardware
//
#include <hip/hip_runtime.h>
#include <hip/hip_bf16.h>

// ---------------------------------------------------------------------------
// DMFM-Lite forward for MI455X (gfx1250, wave32, WMMA).
// Dense GEMMs: v_wmma_f32_16x16x32_bf16, LDS staged via CDNA5 async
// global->LDS copies (global_load_async_to_lds_b128 / s_wait_asynccnt),
// b128 LDS fragment loads. GAT segment-softmax: 3-pass global atomics.
// ---------------------------------------------------------------------------

typedef __bf16 bf16_t;
typedef __attribute__((ext_vector_type(16))) __bf16 v16bf;
typedef __attribute__((ext_vector_type(8)))  float  v8f;
typedef __attribute__((ext_vector_type(4)))  int    v4i;

#define NN      30000
#define FF      128
#define HID     64
#define HEADS   2
#define E_IND   960000
#define E_UNI   480000
#define BN_EPS  1e-5f

// ---------------------------------------------------------------------------
// CDNA5 async global->LDS copy (ISA 15.18.3, opcode 98; tracked by ASYNCcnt)
// ---------------------------------------------------------------------------
__device__ __forceinline__ void async_ld_b128(void* lds_dst, const void* g_src) {
    // LDS aperture maps addr[31:0] directly -> low 32 bits of generic pointer
    unsigned l = (unsigned)(uintptr_t)lds_dst;
    asm volatile("global_load_async_to_lds_b128 %0, %1, off"
                 :: "v"(l), "v"(g_src)
                 : "memory");
}
__device__ __forceinline__ void wait_async0() {
    asm volatile("s_wait_asynccnt 0" ::: "memory");
}

// ---------------------------------------------------------------------------
// small helpers
// ---------------------------------------------------------------------------
__device__ __forceinline__ void atomicMaxF(float* addr, float v) {
    if (v >= 0.0f) atomicMax((int*)addr, __float_as_int(v));
    else           atomicMin((unsigned int*)addr, __float_as_uint(v));
}

// ---------------------------------------------------------------------------
// zero / init kernels (ws is poisoned; atomics need explicit init each call)
// ---------------------------------------------------------------------------
__global__ void k_zero256(float* p) {
    p[threadIdx.x] = 0.0f;  // 256 threads: sum[128] + sumsq[128]
}

__global__ void k_gat_init(float* out, float* amax, float* den, int nk, int nh) {
    int i = blockIdx.x * blockDim.x + threadIdx.x;
    if (i < nk) out[i] = 0.0f;
    if (i < nh) { amax[i] = __int_as_float(0xFF800000); den[i] = 0.0f; }
}

// ---------------------------------------------------------------------------
// BatchNorm statistics (per-feature over 30000 rows)
// ---------------------------------------------------------------------------
__global__ void k_bn_stats(const float* __restrict__ x, float* sum, float* sq) {
    const int f  = threadIdx.x;            // 128 threads: one feature each
    const int r0 = blockIdx.x * 256;
    const int r1 = min(r0 + 256, NN);
    float s = 0.0f, q = 0.0f;
    for (int r = r0; r < r1; ++r) {
        float v = x[(size_t)r * FF + f];
        s += v; q += v * v;
    }
    atomicAdd(&sum[f], s);
    atomicAdd(&sq[f],  q);
}

__global__ void k_bn_fin(const float* sum, const float* sq,
                         const float* gamma, const float* beta,
                         float* scale, float* shift) {
    const int f = threadIdx.x;             // 128 threads
    float mu  = sum[f] * (1.0f / NN);
    float var = sq[f]  * (1.0f / NN) - mu * mu;
    float a   = gamma[f] * rsqrtf(var + BN_EPS);
    scale[f]  = a;
    shift[f]  = beta[f] - mu * a;
}

// x_norm -> bf16 (for encoder GEMM) and raw x -> bf16 (for attention GEMM)
__global__ void k_bn_apply(const float* __restrict__ x,
                           const float* __restrict__ scale,
                           const float* __restrict__ shift,
                           bf16_t* __restrict__ xn, bf16_t* __restrict__ xb,
                           int total) {
    int i = blockIdx.x * blockDim.x + threadIdx.x;
    if (i >= total) return;
    int f = i & (FF - 1);
    float v = x[i];
    xn[i] = (bf16_t)(v * scale[f] + shift[f]);
    xb[i] = (bf16_t)v;
}

// convert f32 weight [K x N] to bf16 TRANSPOSED [N x K] (for N-major GEMM B)
__global__ void k_f2bf_t(const float* __restrict__ in, bf16_t* __restrict__ out,
                         int K, int N) {
    int i = blockIdx.x * blockDim.x + threadIdx.x;
    if (i >= K * N) return;
    int k = i / N, n = i % N;
    out[(size_t)n * K + k] = (bf16_t)in[i];
}

// ---------------------------------------------------------------------------
// bf16 WMMA GEMM: C[M x N] = act(A[M x KDIM] * BT^T + bias)
//   A : row-major [M, KDIM] bf16
//   BT: N-major   [N, KDIM] bf16 (pre-transposed weight)
// Block = 128 threads = 4 wave32 waves, each wave owns a 16x16 tile
// (block tile 16 x 64). Entire A row-block (16*KDIM) and B tile (64*KDIM)
// staged to LDS via async global->LDS b128 copies; one s_wait_asynccnt +
// barrier; fully unrolled WMMA K-loop with ds_load_b128 fragment fetches.
// ACT: 0 = none, 1 = relu, 2 = leaky-relu(0.2)
// ---------------------------------------------------------------------------
template <int KDIM, int ACT, bool WBF>
__global__ void k_gemm_bf16(const bf16_t* __restrict__ A,
                            const bf16_t* __restrict__ BT,
                            const float*  __restrict__ bias,
                            float*  __restrict__ Co,
                            bf16_t* __restrict__ Cb,
                            int M, int N) {
    __shared__ bf16_t sA[16 * KDIM];   // <= 6 KB
    __shared__ bf16_t sB[64 * KDIM];   // <= 24 KB

    constexpr int NA = 16 * KDIM / 8;  // 16B chunks in A block
    constexpr int NB = 64 * KDIM / 8;  // 16B chunks in B tile
    static_assert(NA % 128 == 0 && NB % 128 == 0, "chunk counts must fill block");

    const int tid = threadIdx.x;
    const int w   = tid >> 5;       // wave 0..3
    const int l   = tid & 31;       // lane
    const int m0  = blockIdx.x * 16;
    const int n0  = blockIdx.y * 64;

    const int row = l & 15;                 // A-fragment row (M)
    const int kb  = (l < 16) ? 0 : 8;       // per-lane K base (ISA 16-bit layout)
    const int cw  = (w << 4) + (l & 15);    // column within 64-wide block tile

    // ---- async stage A block (16 consecutive rows of KDIM, contiguous)
    {
        const v4i* __restrict__ ga = (const v4i*)(A + (size_t)m0 * KDIM);
        v4i* sa = (v4i*)sA;
#pragma unroll
        for (int i = 0; i < NA / 128; ++i)
            async_ld_b128(&sa[i * 128 + tid], &ga[i * 128 + tid]);
        __builtin_prefetch(A + (size_t)(m0 + 16) * KDIM, 0, 1);  // next M block
    }
    // ---- async stage B tile (rows n0..n0+63 of BT, contiguous)
    {
        const v4i* __restrict__ gb = (const v4i*)(BT + (size_t)n0 * KDIM);
        v4i* sb = (v4i*)sB;
#pragma unroll
        for (int i = 0; i < NB / 128; ++i)
            async_ld_b128(&sb[i * 128 + tid], &gb[i * 128 + tid]);
    }
    wait_async0();
    __syncthreads();

    v8f acc = {};
    union Frag { v16bf v; v4i q[2]; };

#pragma unroll
    for (int k0 = 0; k0 < KDIM; k0 += 32) {
        Frag a, b;
        // A fragment: lane (row, kb) -> K runs [k0+kb, +8) and [k0+16+kb, +8)
        a.q[0] = *(const v4i*)&sA[row * KDIM + k0 + kb];
        a.q[1] = *(const v4i*)&sA[row * KDIM + k0 + 16 + kb];
        // B fragment: lane col cw -> same K runs (sB is N-major)
        b.q[0] = *(const v4i*)&sB[cw * KDIM + k0 + kb];
        b.q[1] = *(const v4i*)&sB[cw * KDIM + k0 + 16 + kb];
        acc = __builtin_amdgcn_wmma_f32_16x16x32_bf16(
                  false, a.v, false, b.v, (short)0, acc, false, false);
    }

    const int   nn = n0 + cw;
    const float bb = bias ? bias[nn] : 0.0f;
#pragma unroll
    for (int r = 0; r < 8; ++r) {
        int   m = m0 + r + ((l >> 4) << 3);   // VGPR r: M=r (lanes 0-15), M=r+8 (16-31)
        float v = acc[r] + bb;
        if (ACT == 1)      v = v > 0.0f ? v : 0.0f;
        else if (ACT == 2) v = v > 0.0f ? v : 0.2f * v;
        Co[(size_t)m * N + nn] = v;
        if (WBF) Cb[(size_t)m * N + nn] = (bf16_t)v;
    }
}

// ---------------------------------------------------------------------------
// GAT pieces
// ---------------------------------------------------------------------------
__global__ void k_att_dots(const float* __restrict__ Hm,
                           const float* __restrict__ att_s,
                           const float* __restrict__ att_d,
                           float* __restrict__ asrc, float* __restrict__ adst) {
    int i = blockIdx.x * blockDim.x + threadIdx.x;
    if (i >= NN * HEADS) return;
    int n = i >> 1, h = i & 1;
    const float* hp = Hm + (size_t)n * (HEADS * HID) + h * HID;
    const float* as = att_s + h * HID;
    const float* ad = att_d + h * HID;
    float s = 0.0f, d = 0.0f;
#pragma unroll 4
    for (int c = 0; c < HID; ++c) { float x = hp[c]; s += x * as[c]; d += x * ad[c]; }
    asrc[i] = s; adst[i] = d;
}

__device__ __forceinline__ void edge_sd(const int* ei, int E, int e, int& s, int& d) {
    if (e < E) { s = ei[e]; d = ei[E + e]; }
    else       { s = d = e - E; }           // self loop
}

__global__ void k_edge_max(const int* __restrict__ ei, int E, int Etot,
                           const float* __restrict__ asrc,
                           const float* __restrict__ adst,
                           float* __restrict__ amax) {
    int e = blockIdx.x * blockDim.x + threadIdx.x;
    if (e >= Etot) return;
    int s, d; edge_sd(ei, E, e, s, d);
#pragma unroll
    for (int h = 0; h < HEADS; ++h) {
        float a = asrc[s * HEADS + h] + adst[d * HEADS + h];
        a = a > 0.0f ? a : 0.2f * a;
        atomicMaxF(&amax[d * HEADS + h], a);
    }
}

__global__ void k_edge_den(const int* __restrict__ ei, int E, int Etot,
                           const float* __restrict__ asrc,
                           const float* __restrict__ adst,
                           const float* __restrict__ amax,
                           float* __restrict__ den) {
    int e = blockIdx.x * blockDim.x + threadIdx.x;
    if (e >= Etot) return;
    int s, d; edge_sd(ei, E, e, s, d);
#pragma unroll
    for (int h = 0; h < HEADS; ++h) {
        float a = asrc[s * HEADS + h] + adst[d * HEADS + h];
        a = a > 0.0f ? a : 0.2f * a;
        atomicAdd(&den[d * HEADS + h], __expf(a - amax[d * HEADS + h]));
    }
}

__global__ void k_edge_scatter(const int* __restrict__ ei, int E, int Etot,
                               const float* __restrict__ asrc,
                               const float* __restrict__ adst,
                               const float* __restrict__ amax,
                               const float* __restrict__ den,
                               const float* __restrict__ Hm,
                               float* __restrict__ out) {
    int idx = blockIdx.x * blockDim.x + threadIdx.x;
    if (idx >= Etot * HEADS) return;
    int e = idx >> 1, h = idx & 1;
    int s, d; edge_sd(ei, E, e, s, d);
    float a = asrc[s * HEADS + h] + adst[d * HEADS + h];
    a = a > 0.0f ? a : 0.2f * a;
    float wgt = __expf(a - amax[d * HEADS + h]) / den[d * HEADS + h];
    const float* hs = Hm  + (size_t)s * (HEADS * HID) + h * HID;
    float*       op = out + (size_t)d * (HEADS * HID) + h * HID;
#pragma unroll 4
    for (int c = 0; c < HID; ++c) atomicAdd(&op[c], hs[c] * wgt);
}

__global__ void k_gat_fin(const float* __restrict__ out,
                          const float* __restrict__ bias,
                          const float* __restrict__ Cprev,
                          float* __restrict__ Hout,
                          float* __restrict__ Cnext,
                          bf16_t* __restrict__ CnextB) {
    int i = blockIdx.x * blockDim.x + threadIdx.x;
    if (i >= NN * HID) return;
    int n = i >> 6, c = i & (HID - 1);
    float m = 0.5f * (out[(size_t)n * (HEADS * HID) + c] +
                      out[(size_t)n * (HEADS * HID) + HID + c]) + bias[c];
    float h  = m > 0.0f ? m : (__expf(m) - 1.0f);     // elu
    float ci = Cprev[i] - h;
    Hout[i]  = h;
    Cnext[i] = ci;
    if (CnextB) CnextB[i] = (bf16_t)ci;
}

// ---------------------------------------------------------------------------
// decoder / attention tail
// ---------------------------------------------------------------------------
__global__ void k_build_hier(const float* __restrict__ C,
                             const float* __restrict__ CI,
                             const float* __restrict__ CU,
                             bf16_t* __restrict__ hier) {
    int i = blockIdx.x * blockDim.x + threadIdx.x;
    if (i >= NN * 3 * HID) return;
    int n = i / (3 * HID), j = i % (3 * HID);
    float v = (j < HID) ? C[(size_t)n * HID + j]
            : (j < 2 * HID) ? CI[(size_t)n * HID + j - HID]
                            : CU[(size_t)n * HID + j - 2 * HID];
    hier[i] = (bf16_t)v;
}

__global__ void k_deep(const float* __restrict__ D1,
                       const float* __restrict__ Wd2,
                       const float* __restrict__ bd2,
                       float* __restrict__ deep) {
    __shared__ float red[HID];
    int n = blockIdx.x, t = threadIdx.x;
    red[t] = D1[(size_t)n * HID + t] * Wd2[t];
    __syncthreads();
    for (int s = HID / 2; s > 0; s >>= 1) {
        if (t < s) red[t] += red[t + s];
        __syncthreads();
    }
    if (t == 0) deep[n] = red[0] + bd2[0];
}

__global__ void k_softmax(const float* __restrict__ U, float* __restrict__ attn) {
    __shared__ float red[FF];
    int n = blockIdx.x, t = threadIdx.x;
    float u = U[(size_t)n * FF + t];
    red[t] = u; __syncthreads();
    for (int s = FF / 2; s > 0; s >>= 1) {
        if (t < s) red[t] = fmaxf(red[t], red[t + s]);
        __syncthreads();
    }
    float mx = red[0]; __syncthreads();
    float e = __expf(u - mx);
    red[t] = e; __syncthreads();
    for (int s = FF / 2; s > 0; s >>= 1) {
        if (t < s) red[t] += red[t + s];
        __syncthreads();
    }
    attn[(size_t)n * FF + t] = e / red[0];
}

// ---------------------------------------------------------------------------
// launcher
// ---------------------------------------------------------------------------
extern "C" void kernel_launch(void* const* d_in, const int* in_sizes, int n_in,
                              void* d_out, int out_size, void* d_ws, size_t ws_size,
                              hipStream_t stream) {
    (void)in_sizes; (void)n_in; (void)out_size; (void)ws_size;

    // ---- inputs (setup_inputs dict order) ----
    const float* x        = (const float*)d_in[0];
    const int*   ei1      = (const int*)  d_in[1];
    const int*   ei2      = (const int*)  d_in[2];
    const float* bn_gamma = (const float*)d_in[3];
    const float* bn_beta  = (const float*)d_in[4];
    const float* W_enc    = (const float*)d_in[5];
    const float* b_enc    = (const float*)d_in[6];
    const float* Wg1      = (const float*)d_in[7];
    const float* att_s1   = (const float*)d_in[8];
    const float* att_d1   = (const float*)d_in[9];
    const float* bias1    = (const float*)d_in[10];
    const float* Wg2      = (const float*)d_in[11];
    const float* att_s2   = (const float*)d_in[12];
    const float* att_d2   = (const float*)d_in[13];
    const float* bias2    = (const float*)d_in[14];
    const float* Wd1      = (const float*)d_in[15];
    const float* bd1      = (const float*)d_in[16];
    const float* Wd2      = (const float*)d_in[17];
    const float* bd2      = (const float*)d_in[18];
    const float* Wa       = (const float*)d_in[19];
    const float* ba       = (const float*)d_in[20];

    // ---- output regions (tuple flattened) ----
    float* out  = (float*)d_out;
    float* DEEP = out;                          // [30000]
    float* ATTN = out + 30000;                  // [30000,128]
    float* Cf   = out + 3870000;                // [30000,64]
    float* CIf  = out + 5790000;
    float* CUf  = out + 7710000;
    float* HIf  = out + 9630000;
    float* HUf  = out + 11550000;

    // ---- workspace layout (all bf16 regions 16B-aligned) ----
    char*  ws     = (char*)d_ws;
    float* sumf   = (float*)ws;                 // [128]
    float* sqf    = sumf + 128;                 // [128]
    float* scalef = sumf + 256;                 // [128]
    float* shiftf = sumf + 384;                 // [128]
    bf16_t* WencT = (bf16_t*)(ws + 4096);       // [64,128]  (N x K)
    bf16_t* Wg1T  = WencT + 64 * 128;           // [128,64]
    bf16_t* Wg2T  = Wg1T  + 128 * 64;           // [128,64]
    bf16_t* Wd1T  = Wg2T  + 128 * 64;           // [64,192]
    bf16_t* WaT   = Wd1T  + 64 * 192;           // [128,128]
    bf16_t* XN    = WaT   + 128 * 128;          // [N,128] normalized bf16
    bf16_t* XB    = XN  + (size_t)NN * FF;      // [N,128] raw bf16
    bf16_t* CB    = XB  + (size_t)NN * FF;      // [N,64]
    bf16_t* CIB   = CB  + (size_t)NN * HID;     // [N,64]
    bf16_t* HIER  = CIB + (size_t)NN * HID;     // [N,192]
    float*  Hf    = (float*)(HIER + (size_t)NN * 3 * HID); // [N,128] h / later D1
    float*  OUTf  = Hf   + (size_t)NN * FF;     // [N,128] agg / later U
    float*  ASRC  = OUTf + (size_t)NN * FF;     // [N*2]
    float*  ADST  = ASRC + NN * HEADS;
    float*  AMAX  = ADST + NN * HEADS;
    float*  DEN   = AMAX + NN * HEADS;

    const int T = 256;
    auto cdiv = [](int a, int b) { return (a + b - 1) / b; };

    // ---- BatchNorm (stats -> affine -> apply + bf16 conversions) ----
    k_zero256<<<1, 256, 0, stream>>>(sumf);
    k_bn_stats<<<cdiv(NN, 256), 128, 0, stream>>>(x, sumf, sqf);
    k_bn_fin<<<1, 128, 0, stream>>>(sumf, sqf, bn_gamma, bn_beta, scalef, shiftf);
    k_bn_apply<<<cdiv(NN * FF, T), T, 0, stream>>>(x, scalef, shiftf, XN, XB, NN * FF);

    // ---- weight conversion: f32 [K,N] -> bf16 transposed [N,K] ----
    k_f2bf_t<<<cdiv(128 * 64, T), T, 0, stream>>>(W_enc, WencT, 128, 64);
    k_f2bf_t<<<cdiv(64 * 128, T), T, 0, stream>>>(Wg1, Wg1T, 64, 128);
    k_f2bf_t<<<cdiv(64 * 128, T), T, 0, stream>>>(Wg2, Wg2T, 64, 128);
    k_f2bf_t<<<cdiv(192 * 64, T), T, 0, stream>>>(Wd1, Wd1T, 192, 64);
    k_f2bf_t<<<cdiv(128 * 128, T), T, 0, stream>>>(Wa, WaT, 128, 128);

    // ---- encoder: C = relu(x_norm @ W_enc + b_enc)  [WMMA, K=128] ----
    k_gemm_bf16<128, 1, true><<<dim3(NN / 16, 1), 128, 0, stream>>>(
        XN, WencT, b_enc, Cf, CB, NN, HID);

    // ---- GAT conv 1 (industry) ----
    {
        const int E = E_IND, Etot = E_IND + NN;
        k_gemm_bf16<64, 0, false><<<dim3(NN / 16, 2), 128, 0, stream>>>(
            CB, Wg1T, nullptr, Hf, nullptr, NN, HEADS * HID);
        k_gat_init<<<cdiv(NN * FF, T), T, 0, stream>>>(OUTf, AMAX, DEN, NN * FF, NN * HEADS);
        k_att_dots<<<cdiv(NN * HEADS, T), T, 0, stream>>>(Hf, att_s1, att_d1, ASRC, ADST);
        k_edge_max<<<cdiv(Etot, T), T, 0, stream>>>(ei1, E, Etot, ASRC, ADST, AMAX);
        k_edge_den<<<cdiv(Etot, T), T, 0, stream>>>(ei1, E, Etot, ASRC, ADST, AMAX, DEN);
        k_edge_scatter<<<cdiv(Etot * HEADS, T), T, 0, stream>>>(
            ei1, E, Etot, ASRC, ADST, AMAX, DEN, Hf, OUTf);
        k_gat_fin<<<cdiv(NN * HID, T), T, 0, stream>>>(OUTf, bias1, Cf, HIf, CIf, CIB);
    }

    // ---- GAT conv 2 (universe) ----
    {
        const int E = E_UNI, Etot = E_UNI + NN;
        k_gemm_bf16<64, 0, false><<<dim3(NN / 16, 2), 128, 0, stream>>>(
            CIB, Wg2T, nullptr, Hf, nullptr, NN, HEADS * HID);
        k_gat_init<<<cdiv(NN * FF, T), T, 0, stream>>>(OUTf, AMAX, DEN, NN * FF, NN * HEADS);
        k_att_dots<<<cdiv(NN * HEADS, T), T, 0, stream>>>(Hf, att_s2, att_d2, ASRC, ADST);
        k_edge_max<<<cdiv(Etot, T), T, 0, stream>>>(ei2, E, Etot, ASRC, ADST, AMAX);
        k_edge_den<<<cdiv(Etot, T), T, 0, stream>>>(ei2, E, Etot, ASRC, ADST, AMAX, DEN);
        k_edge_scatter<<<cdiv(Etot * HEADS, T), T, 0, stream>>>(
            ei2, E, Etot, ASRC, ADST, AMAX, DEN, Hf, OUTf);
        k_gat_fin<<<cdiv(NN * HID, T), T, 0, stream>>>(OUTf, bias2, CIf, HUf, CUf, nullptr);
    }

    // ---- decoder: hier=[C,C_I,C_U]; D1=relu(hier@Wd1+bd1) [WMMA K=192]; deep=D1@Wd2+bd2 ----
    k_build_hier<<<cdiv(NN * 3 * HID, T), T, 0, stream>>>(Cf, CIf, CUf, HIER);
    k_gemm_bf16<192, 1, false><<<dim3(NN / 16, 1), 128, 0, stream>>>(
        HIER, Wd1T, bd1, Hf /*reuse as D1*/, nullptr, NN, HID);
    k_deep<<<NN, HID, 0, stream>>>(Hf, Wd2, bd2, DEEP);

    // ---- factor attention: U = leaky_relu(x @ Wa + ba) [WMMA K=128]; softmax ----
    k_gemm_bf16<128, 2, false><<<dim3(NN / 16, 2), 128, 0, stream>>>(
        XB, WaT, ba, OUTf /*reuse as U*/, nullptr, NN, FF);
    k_softmax<<<NN, FF, 0, stream>>>(OUTf, ATTN);
}
